// SwinTransformer_73065983640131
// MI455X (gfx1250) — matrix operand order repaired and checked by reading, not compiled, not run on hardware
//
#include <hip/hip_runtime.h>
#include <hip/hip_bf16.h>
#include <math.h>

typedef __attribute__((ext_vector_type(16))) __bf16 v16bf;
typedef __attribute__((ext_vector_type(8)))  float  v8f;

// ---------------------------------------------------------------------------
// Weight fp32 -> bf16 conversion (per-GEMM staging into L2-resident scratch)
// ---------------------------------------------------------------------------
__global__ void wcvt_kernel(const float* __restrict__ in, __bf16* __restrict__ out, int total)
{
  int idx = blockIdx.x * blockDim.x + threadIdx.x;
  if (idx < total) out[idx] = (__bf16)in[idx];
}

// ---------------------------------------------------------------------------
// GEMM: out[M,N] = act(A[M,K] @ W[N,K]^T + bias) (+ res)
// A fp32 (converted in-flight), W pre-converted bf16. WMMA 16x16x32 bf16,
// fp32 accumulate. 8 waves/block, wave tile 32x64 (2 A-frags x 4 B-frags).
// NOTE: res/out may alias (in-place residual) -> no restrict on them.
// ---------------------------------------------------------------------------
__global__ __launch_bounds__(256) void gemm_kernel(
    const float* __restrict__ A, const __bf16* __restrict__ Wb,
    const float* __restrict__ bias, const float* res,
    float* out, int M, int N, int K, int act)
{
  const int lane = threadIdx.x & 31;
  const int wave = threadIdx.x >> 5;
  const int ln   = lane & 15;
  const int hi   = lane >> 4;
  const int rbase = blockIdx.x * 256 + wave * 32;
  const int cbase = blockIdx.y * 64;

  int arow0 = rbase + ln;       if (arow0 >= M) arow0 = M - 1;
  int arow1 = rbase + 16 + ln;  if (arow1 >= M) arow1 = M - 1;
  const float* ap0 = A + (size_t)arow0 * K + hi * 16;
  const float* ap1 = A + (size_t)arow1 * K + hi * 16;

  v8f acc0[4] = {};
  v8f acc1[4] = {};
  for (int kb = 0; kb < K; kb += 32) {
    v16bf a0, a1;
#pragma unroll
    for (int e = 0; e < 8; ++e) {
      a0[e]     = (__bf16)ap0[kb + e];
      a0[8 + e] = (__bf16)ap0[kb + 8 + e];
      a1[e]     = (__bf16)ap1[kb + e];
      a1[8 + e] = (__bf16)ap1[kb + 8 + e];
    }
#pragma unroll
    for (int j = 0; j < 4; ++j) {
      int col = cbase + 16 * j + ln;
      int colc = col < N ? col : N - 1;        // clamp for edge N-tiles
      const __bf16* bp = Wb + (size_t)colc * K + kb + hi * 16;
      v16bf b;
#pragma unroll
      for (int e = 0; e < 16; ++e) b[e] = bp[e];
      acc0[j] = __builtin_amdgcn_wmma_f32_16x16x32_bf16(
          false, a0, false, b, (short)0, acc0[j], false, false);
      acc1[j] = __builtin_amdgcn_wmma_f32_16x16x32_bf16(
          false, a1, false, b, (short)0, acc1[j], false, false);
    }
  }

#pragma unroll
  for (int j = 0; j < 4; ++j) {
    int col = cbase + 16 * j + ln;
    if (col >= N) continue;
    float bv = bias ? bias[col] : 0.f;
#pragma unroll
    for (int r = 0; r < 8; ++r) {
      int row0 = rbase + r + 8 * hi;
      if (row0 < M) {
        float v = acc0[j][r] + bv;
        if (act) v = 0.5f * v * (1.f + erff(v * 0.70710678118f));
        if (res) v += res[(size_t)row0 * N + col];
        out[(size_t)row0 * N + col] = v;
      }
      int row1 = rbase + 16 + r + 8 * hi;
      if (row1 < M) {
        float v = acc1[j][r] + bv;
        if (act) v = 0.5f * v * (1.f + erff(v * 0.70710678118f));
        if (res) v += res[(size_t)row1 * N + col];
        out[(size_t)row1 * N + col] = v;
      }
    }
  }
}

// ---------------------------------------------------------------------------
// Window attention: one block (4 waves, 128 thr) per (window, head). d = 32.
// ---------------------------------------------------------------------------
__global__ __launch_bounds__(128) void attn_kernel(
    const float* __restrict__ qkv, const float* __restrict__ rpb,
    float* __restrict__ out, int heads, int C, int H, int W, int shiftFlag)
{
  const int nWw = W / 7;
  const int nW  = (H / 7) * nWw;
  const int h   = blockIdx.x % heads;
  const int win = blockIdx.x / heads;
  const int wi  = win % nW;
  const int wrow = wi / nWw, wcol = wi % nWw;

  const int tid  = threadIdx.x;
  const int lane = tid & 31;
  const int wave = tid >> 5;
  const int ln   = lane & 15;
  const int hi   = lane >> 4;

  __shared__ __bf16 qs[64 * 32];
  __shared__ __bf16 ks[64 * 32];
  __shared__ __bf16 vts[32 * 64];
  __shared__ float  Smat[64 * 64];
  __shared__ __bf16 Pmat[64 * 64];

  for (int i = tid; i < 64 * 32; i += 128) {
    qs[i] = (__bf16)0.f; ks[i] = (__bf16)0.f; vts[i] = (__bf16)0.f;
  }
  __syncthreads();

  const float scale = 0.17677669529f;  // 32^-0.5
  const size_t base = (size_t)win * 49 * (3 * C) + (size_t)h * 32;
  for (int i = tid; i < 49 * 32; i += 128) {
    int t = i >> 5, di = i & 31;
    size_t r = base + (size_t)t * 3 * C + di;
    qs[t * 32 + di]  = (__bf16)(qkv[r] * scale);
    ks[t * 32 + di]  = (__bf16)(qkv[r + C]);
    vts[di * 64 + t] = (__bf16)(qkv[r + 2 * C]);
  }
  __syncthreads();

  const int rbase = wave * 16;

  // S = q @ k^T  (K = 32, single step)
  v8f accS[4] = {};
  {
    v16bf a;
    const __bf16* qp = qs + (rbase + ln) * 32 + hi * 16;
#pragma unroll
    for (int e = 0; e < 8; ++e) { a[e] = qp[e]; a[8 + e] = qp[8 + e]; }
#pragma unroll
    for (int j = 0; j < 4; ++j) {
      v16bf b;
      const __bf16* kp = ks + (16 * j + ln) * 32 + hi * 16;
#pragma unroll
      for (int e = 0; e < 16; ++e) b[e] = kp[e];
      accS[j] = __builtin_amdgcn_wmma_f32_16x16x32_bf16(
          false, a, false, b, (short)0, accS[j], false, false);
    }
  }

  // Write S to LDS with rel-pos bias and (optional) shift mask
#pragma unroll
  for (int j = 0; j < 4; ++j) {
    int ct = 16 * j + ln;
#pragma unroll
    for (int r = 0; r < 8; ++r) {
      int rt = rbase + r + 8 * hi;
      float sv = accS[j][r];
      if (rt < 49 && ct < 49) {
        int ih = rt / 7, iw = rt - ih * 7;
        int jh = ct / 7, jw = ct - jh * 7;
        sv += rpb[((ih - jh + 6) * 13 + (iw - jw + 6)) * heads + h];
        if (shiftFlag) {
          int hiI = wrow * 7 + ih, wiI = wcol * 7 + iw;
          int hiJ = wrow * 7 + jh, wiJ = wcol * 7 + jw;
          int ri = (hiI < H - 7 ? 0 : (hiI < H - 3 ? 1 : 2)) * 3 +
                   (wiI < W - 7 ? 0 : (wiI < W - 3 ? 1 : 2));
          int rj = (hiJ < H - 7 ? 0 : (hiJ < H - 3 ? 1 : 2)) * 3 +
                   (wiJ < W - 7 ? 0 : (wiJ < W - 3 ? 1 : 2));
          if (ri != rj) sv -= 100.f;
        }
      }
      Smat[rt * 64 + ct] = sv;
    }
  }
  __syncthreads();

  // Row softmax (rows 0..48 valid; pad rows/cols -> 0 in P)
  if (tid < 64) {
    float* sr = Smat + tid * 64;
    __bf16* pr = Pmat + tid * 64;
    if (tid < 49) {
      float mx = -1e30f;
      for (int c = 0; c < 49; ++c) mx = fmaxf(mx, sr[c]);
      float sum = 0.f;
      for (int c = 0; c < 49; ++c) sum += expf(sr[c] - mx);
      float rinv = 1.f / sum;
      for (int c = 0; c < 49; ++c) pr[c] = (__bf16)(expf(sr[c] - mx) * rinv);
      for (int c = 49; c < 64; ++c) pr[c] = (__bf16)0.f;
    } else {
      for (int c = 0; c < 64; ++c) pr[c] = (__bf16)0.f;
    }
  }
  __syncthreads();

  // O = P @ V  (K = 64, two steps; N = 32 -> 2 frags)
  v8f accO[2] = {};
#pragma unroll
  for (int kb = 0; kb < 64; kb += 32) {
    v16bf a;
    const __bf16* pp = Pmat + (rbase + ln) * 64 + kb + hi * 16;
#pragma unroll
    for (int e = 0; e < 8; ++e) { a[e] = pp[e]; a[8 + e] = pp[8 + e]; }
#pragma unroll
    for (int j = 0; j < 2; ++j) {
      v16bf b;
      const __bf16* vp = vts + (16 * j + ln) * 64 + kb + hi * 16;
#pragma unroll
      for (int e = 0; e < 16; ++e) b[e] = vp[e];
      accO[j] = __builtin_amdgcn_wmma_f32_16x16x32_bf16(
          false, a, false, b, (short)0, accO[j], false, false);
    }
  }

#pragma unroll
  for (int j = 0; j < 2; ++j) {
    int col = h * 32 + 16 * j + ln;
#pragma unroll
    for (int r = 0; r < 8; ++r) {
      int rt = rbase + r + 8 * hi;
      if (rt < 49) out[((size_t)win * 49 + rt) * C + col] = accO[j][r];
    }
  }
}

// ---------------------------------------------------------------------------
// LayerNorm: one block (128 thr) per row.
// ---------------------------------------------------------------------------
__global__ __launch_bounds__(128) void ln_kernel(
    const float* __restrict__ x, const float* __restrict__ g,
    const float* __restrict__ b, float* __restrict__ y, int C)
{
  const float* xr = x + (size_t)blockIdx.x * C;
  float s = 0.f, s2 = 0.f;
  for (int c = threadIdx.x; c < C; c += 128) { float v = xr[c]; s += v; s2 += v * v; }
  __shared__ float rs[128], rq[128];
  rs[threadIdx.x] = s; rq[threadIdx.x] = s2; __syncthreads();
  for (int off = 64; off > 0; off >>= 1) {
    if (threadIdx.x < off) { rs[threadIdx.x] += rs[threadIdx.x + off];
                             rq[threadIdx.x] += rq[threadIdx.x + off]; }
    __syncthreads();
  }
  float mean = rs[0] / C;
  float var  = rq[0] / C - mean * mean;
  float inv  = rsqrtf(var + 1e-6f);
  float* yr = y + (size_t)blockIdx.x * C;
  for (int c = threadIdx.x; c < C; c += 128)
    yr[c] = (xr[c] - mean) * inv * g[c] + b[c];
}

// LayerNorm + NHWC -> NCHW stage output writer
__global__ __launch_bounds__(128) void ln_chw_kernel(
    const float* __restrict__ x, const float* __restrict__ g,
    const float* __restrict__ b, float* __restrict__ y, int C, int HW)
{
  const int row = blockIdx.x;            // b*HW + l
  const int bb = row / HW, l = row - bb * HW;
  const float* xr = x + (size_t)row * C;
  float s = 0.f, s2 = 0.f;
  for (int c = threadIdx.x; c < C; c += 128) { float v = xr[c]; s += v; s2 += v * v; }
  __shared__ float rs[128], rq[128];
  rs[threadIdx.x] = s; rq[threadIdx.x] = s2; __syncthreads();
  for (int off = 64; off > 0; off >>= 1) {
    if (threadIdx.x < off) { rs[threadIdx.x] += rs[threadIdx.x + off];
                             rq[threadIdx.x] += rq[threadIdx.x + off]; }
    __syncthreads();
  }
  float mean = rs[0] / C;
  float var  = rq[0] / C - mean * mean;
  float inv  = rsqrtf(var + 1e-6f);
  for (int c = threadIdx.x; c < C; c += 128)
    y[((size_t)bb * C + c) * HW + l] = (xr[c] - mean) * inv * g[c] + b[c];
}

// ---------------------------------------------------------------------------
// Patch embed: 4x4/4 conv, NCHW in -> [B, 56*56, 96] out
// ---------------------------------------------------------------------------
__global__ void patch_embed_kernel(
    const float* __restrict__ x, const float* __restrict__ w,
    const float* __restrict__ bias, float* __restrict__ y, int B)
{
  int idx = blockIdx.x * blockDim.x + threadIdx.x;
  const int total = B * 3136 * 96;
  if (idx >= total) return;
  int c = idx % 96;
  int l = (idx / 96) % 3136;
  int b = idx / (96 * 3136);
  int oh = l / 56, ow = l % 56;
  float acc = bias[c];
  for (int ci = 0; ci < 3; ++ci)
    for (int kh = 0; kh < 4; ++kh)
      for (int kw = 0; kw < 4; ++kw)
        acc += x[(((size_t)b * 3 + ci) * 224 + oh * 4 + kh) * 224 + ow * 4 + kw] *
               w[((c * 3 + ci) * 4 + kh) * 4 + kw];
  y[idx] = acc;
}

// ---------------------------------------------------------------------------
// Window partition (roll folded in): [B,H,W,C] -> [(B*nW)*49, C]
// ---------------------------------------------------------------------------
__global__ void win_part_kernel(const float* __restrict__ x, float* __restrict__ out,
                                int H, int W, int C, int shift, int total)
{
  int idx = blockIdx.x * blockDim.x + threadIdx.x;
  if (idx >= total) return;
  int c = idx % C;
  int t = (idx / C) % 49;
  int win = idx / (C * 49);
  int nWw = W / 7, nW = (H / 7) * nWw;
  int b = win / nW, wi = win % nW;
  int hh = (wi / nWw) * 7 + t / 7;
  int ww = (wi % nWw) * 7 + t % 7;
  int sh = (hh + shift) % H;
  int sw = (ww + shift) % W;
  out[idx] = x[((size_t)b * H * W + sh * W + sw) * C + c];
}

// Window reverse (+un-roll) with residual add into x (p and x never alias)
__global__ void win_unpart_add_kernel(const float* __restrict__ p, float* __restrict__ x,
                                      int H, int W, int C, int shift, int total)
{
  int idx = blockIdx.x * blockDim.x + threadIdx.x;
  if (idx >= total) return;
  int c = idx % C;
  int l = (idx / C) % (H * W);
  int b = idx / (C * H * W);
  int hh = l / W, ww = l % W;
  int hp = (hh - shift + H) % H;
  int wp = (ww - shift + W) % W;
  int nWw = W / 7, nW = (H / 7) * nWw;
  int wi = (hp / 7) * nWw + wp / 7;
  int t  = (hp % 7) * 7 + wp % 7;
  x[idx] += p[(((size_t)b * nW + wi) * 49 + t) * C + c];
}

// Patch merging gather: [B,H,W,C] -> [B, H/2*W/2, 4C]  (cc = c*4 + hh2*2 + ww2)
__global__ void merge_gather_kernel(const float* __restrict__ x, float* __restrict__ out,
                                    int H, int W, int C, int total)
{
  int idx = blockIdx.x * blockDim.x + threadIdx.x;
  if (idx >= total) return;
  int C4 = 4 * C;
  int H2 = H / 2, W2 = W / 2;
  int cc = idx % C4;
  int l2 = (idx / C4) % (H2 * W2);
  int b  = idx / (C4 * H2 * W2);
  int c = cc >> 2, hh2 = (cc >> 1) & 1, ww2 = cc & 1;
  int h2 = l2 / W2, w2 = l2 % W2;
  out[idx] = x[((size_t)b * H * W + (2 * h2 + hh2) * W + (2 * w2 + ww2)) * C + c];
}

// ---------------------------------------------------------------------------
// Host orchestration
// ---------------------------------------------------------------------------
struct BlockP {
  const float *ln1_g, *ln1_b, *qkv_w, *qkv_b, *rpb, *proj_w, *proj_b,
              *ln2_g, *ln2_b, *fc1_w, *fc1_b, *fc2_w, *fc2_b;
};

extern "C" void kernel_launch(void* const* d_in, const int* in_sizes, int n_in,
                              void* d_out, int out_size, void* d_ws, size_t ws_size,
                              hipStream_t stream)
{
  static const int DEP[4]   = {2, 2, 6, 2};
  static const int HEADS[4] = {3, 6, 12, 24};
  const int B = 8;

  const float* x;
  const float *pe_w, *pe_b, *pe_lg, *pe_lb;
  BlockP blk[12];
  const float *dn_g[3], *dn_b[3], *dn_w[3];
  const float *nr_g[4], *nr_b[4];

  int p = 0;
  auto nx = [&]() { return (const float*)d_in[p++]; };

  if (n_in > 0 && in_sizes[0] == 8 * 3 * 224 * 224) {
    // dict insertion order, depth-first
    x = nx();
    pe_w = nx(); pe_b = nx(); pe_lg = nx(); pe_lb = nx();
    int bi = 0;
    for (int i = 0; i < 4; ++i) {
      for (int j = 0; j < DEP[i]; ++j) {
        BlockP& bp = blk[bi++];
        bp.ln1_g = nx(); bp.ln1_b = nx();
        bp.qkv_w = nx(); bp.qkv_b = nx(); bp.rpb = nx();
        bp.proj_w = nx(); bp.proj_b = nx();
        bp.ln2_g = nx(); bp.ln2_b = nx();
        bp.fc1_w = nx(); bp.fc1_b = nx();
        bp.fc2_w = nx(); bp.fc2_b = nx();
      }
      if (i < 3) { dn_g[i] = nx(); dn_b[i] = nx(); dn_w[i] = nx(); }
    }
    for (int i = 0; i < 4; ++i) { nr_g[i] = nx(); nr_b[i] = nx(); }
  } else {
    // jax tree_flatten order: dict keys sorted, lists in order; x last
    for (int i = 0; i < 4; ++i) { nr_b[i] = nx(); nr_g[i] = nx(); }
    pe_b = nx(); pe_lb = nx(); pe_lg = nx(); pe_w = nx();
    int bi = 0;
    for (int i = 0; i < 4; ++i) {
      for (int j = 0; j < DEP[i]; ++j) {
        BlockP& bp = blk[bi++];
        bp.fc1_b = nx(); bp.fc1_w = nx();
        bp.fc2_b = nx(); bp.fc2_w = nx();
        bp.ln1_b = nx(); bp.ln1_g = nx();
        bp.ln2_b = nx(); bp.ln2_g = nx();
        bp.proj_b = nx(); bp.proj_w = nx();
        bp.qkv_b = nx(); bp.qkv_w = nx();
        bp.rpb = nx();
      }
      if (i < 3) { dn_b[i] = nx(); dn_g[i] = nx(); dn_w[i] = nx(); }
    }
    x = nx();
  }

  // Workspace layout (float units): xb | yb | wb | big | wbf16 scratch
  const size_t SMALL = 2408448;          // 8*3136*96
  const size_t BIG   = 9633792;          // 8*3136*384 (fc1 stage 1)
  const size_t WCVT  = 1179648;          // 3072*768 bf16 elems = 1179648 floats
  float* ws  = (float*)d_ws;
  float* xb  = ws;
  float* yb  = ws + SMALL;
  float* wb  = ws + 2 * SMALL;           // windows / attn-out / merge buffer
  float* big = ws + 3 * SMALL;           // qkv / proj-out / fc1-out
  __bf16* wbf = (__bf16*)(ws + 3 * SMALL + BIG);
  (void)ws_size; (void)out_size; (void)WCVT;

  // helper: stage weight to bf16 then GEMM
  auto gemm = [&](const float* Agem, const float* Wgem, const float* bias,
                  const float* res, float* outp, int M, int N, int K, int act) {
    int wtot = N * K;
    wcvt_kernel<<<(wtot + 255) / 256, 256, 0, stream>>>(Wgem, wbf, wtot);
    dim3 g((M + 255) / 256, (N + 63) / 64);
    gemm_kernel<<<g, 256, 0, stream>>>(Agem, wbf, bias, res, outp, M, N, K, act);
  };

  // ---- patch embed ----
  {
    int total = B * 3136 * 96;
    patch_embed_kernel<<<(total + 255) / 256, 256, 0, stream>>>(x, pe_w, pe_b, yb, B);
    ln_kernel<<<B * 3136, 128, 0, stream>>>(yb, pe_lg, pe_lb, xb, 96);
  }

  int H = 56, W = 56, C = 96;
  float* outf = (float*)d_out;
  size_t out_off = 0;
  int bi = 0;

  for (int i = 0; i < 4; ++i) {
    const int heads = HEADS[i];
    const int L  = H * W;
    const int nW = (H / 7) * (W / 7);
    const int Bn = B * nW;
    const int M  = Bn * 49;              // tokens across all windows

    for (int j = 0; j < DEP[i]; ++j) {
      const BlockP& bp = blk[bi++];
      const int shift = (j % 2 == 0) ? 0 : 3;

      // MSA branch
      ln_kernel<<<B * L, 128, 0, stream>>>(xb, bp.ln1_g, bp.ln1_b, yb, C);
      {
        int total = M * C;
        win_part_kernel<<<(total + 255) / 256, 256, 0, stream>>>(yb, wb, H, W, C, shift, total);
      }
      gemm(wb, bp.qkv_w, bp.qkv_b, nullptr, big, M, 3 * C, C, 0);
      attn_kernel<<<Bn * heads, 128, 0, stream>>>(big, bp.rpb, wb, heads, C, H, W, shift);
      gemm(wb, bp.proj_w, bp.proj_b, nullptr, big, M, C, C, 0);
      {
        int total = B * L * C;
        win_unpart_add_kernel<<<(total + 255) / 256, 256, 0, stream>>>(big, xb, H, W, C, shift, total);
      }

      // MLP branch
      ln_kernel<<<B * L, 128, 0, stream>>>(xb, bp.ln2_g, bp.ln2_b, yb, C);
      gemm(yb, bp.fc1_w, bp.fc1_b, nullptr, big, B * L, 4 * C, C, 1 /*gelu*/);
      gemm(big, bp.fc2_w, bp.fc2_b, xb, xb, B * L, C, 4 * C, 0);
    }

    // Stage output: LN + NHWC->NCHW
    ln_chw_kernel<<<B * L, 128, 0, stream>>>(xb, nr_g[i], nr_b[i], outf + out_off, C, L);
    out_off += (size_t)B * C * L;

    // Patch merging
    if (i < 3) {
      int L4 = L / 4;
      {
        int total = B * L4 * 4 * C;
        merge_gather_kernel<<<(total + 255) / 256, 256, 0, stream>>>(xb, wb, H, W, C, total);
      }
      ln_kernel<<<B * L4, 128, 0, stream>>>(wb, dn_g[i], dn_b[i], yb, 4 * C);
      gemm(yb, dn_w[i], nullptr, nullptr, xb, B * L4, 2 * C, 4 * C, 0);
      H /= 2; W /= 2; C *= 2;
    }
  }
}
